// PygModel_52003464020165
// MI455X (gfx1250) — matrix operand order repaired and checked by reading, not compiled, not run on hardware
//
#include <hip/hip_runtime.h>
#include <hip/hip_bf16.h>

// ---------------- problem constants (match reference) ----------------
#define N_  100000
#define E_  1600000
#define F_  64
#define H_  128
#define R_  20
#define G_  64
#define C_  10
#define L_  2
#define EPS_ 1e-5f

typedef __attribute__((ext_vector_type(16))) _Float16 v16h;
typedef __attribute__((ext_vector_type(8)))  _Float16 v8h;
typedef __attribute__((ext_vector_type(8)))  float    v8f;

// =====================================================================
// WMMA GEMM building block (v_wmma_f32_16x16x32_f16, f32 accumulation)
//
// Block: 256 threads = 8 waves; block tile = 64 rows x 128 cols.
//   wave w: row tile rt = w&3 (16 rows), col half ch = w>>2 (64 cols),
//   one A fragment feeds 4 WMMAs (4 col tiles of 16).
//
// W (f32 in memory) is staged in LDS pre-transposed into the B-fragment
// layout: sW[c*CS + k] = (f16)W[k*128 + c], CS = K + 8 (bank spread).
// A B fragment (16 consecutive k of one column) is then ONE aligned
// 32-byte LDS vector load (2x ds_load_b128) per v_wmma.
// =====================================================================

// ---- input GEMM: A f32 [M,64]  ->  hh f16 [M,128], fused bias+ReLU ----
__global__ __launch_bounds__(256)
void wmma_gemm_in(const float* __restrict__ A, const float* __restrict__ W,
                  const float* __restrict__ bias, _Float16* __restrict__ Ch,
                  int M)
{
    constexpr int K  = F_;
    constexpr int CS = K + 8;
    __shared__ _Float16 sW[128 * CS];

    const int tid  = threadIdx.x;
    const int lane = tid & 31;
    const int wave = tid >> 5;

    for (int i = tid; i < K * 128; i += 256) {
        const int k = i >> 7, n = i & 127;
        sW[n * CS + k] = (_Float16)W[i];
    }
    __syncthreads();

    const int rt = wave & 3;
    const int ch = wave >> 2;
    const int r0 = blockIdx.x * 64 + rt * 16;

    const int mrow  = lane & 15;
    const int khalf = lane >> 4;

    int arow = r0 + mrow;
    if (arow >= M) arow = M - 1;           // OOB rows -> never stored
    const float* __restrict__ Arow = A + (size_t)arow * K;
    __builtin_prefetch(Arow, 0, 0);        // global_prefetch_b8

    const int bcol = lane & 15;
    const _Float16* __restrict__ sWlane = sW + khalf * 16;

    v8f acc[4] = {};

    #pragma unroll
    for (int kc = 0; kc < K; kc += 32) {
        const int kb = kc + khalf * 16;
        const float4 x0 = *(const float4*)(Arow + kb);
        const float4 x1 = *(const float4*)(Arow + kb + 4);
        const float4 x2 = *(const float4*)(Arow + kb + 8);
        const float4 x3 = *(const float4*)(Arow + kb + 12);
        v16h a;
        a[0]=(_Float16)x0.x; a[1]=(_Float16)x0.y; a[2]=(_Float16)x0.z; a[3]=(_Float16)x0.w;
        a[4]=(_Float16)x1.x; a[5]=(_Float16)x1.y; a[6]=(_Float16)x1.z; a[7]=(_Float16)x1.w;
        a[8]=(_Float16)x2.x; a[9]=(_Float16)x2.y; a[10]=(_Float16)x2.z; a[11]=(_Float16)x2.w;
        a[12]=(_Float16)x3.x; a[13]=(_Float16)x3.y; a[14]=(_Float16)x3.z; a[15]=(_Float16)x3.w;

        #pragma unroll
        for (int ct = 0; ct < 4; ++ct) {
            const int c0 = ch * 64 + ct * 16 + bcol;
            const v16h b = *(const v16h*)(sWlane + (size_t)c0 * CS + kc);
            acc[ct] = __builtin_amdgcn_wmma_f32_16x16x32_f16(
                false, a, false, b, (short)0, acc[ct], false, false);
        }
    }

    const int rowOff = (lane < 16) ? 0 : 8;
    #pragma unroll
    for (int ct = 0; ct < 4; ++ct) {
        const int col = ch * 64 + ct * 16 + (lane & 15);
        const float bv = bias[col];
        #pragma unroll
        for (int i = 0; i < 8; ++i) {
            const int row = r0 + i + rowOff;
            if (row < M)
                Ch[(size_t)row * 128 + col] = (_Float16)fmaxf(acc[ct][i] + bv, 0.0f);
        }
    }
}

// ---- layer GEMM: A f16 [M,128] -> C f32 [M,128] (+optional bias) ----
// A fragment = two aligned 16B global loads (no conversion instructions).
__global__ __launch_bounds__(256)
void wmma_gemm_h(const _Float16* __restrict__ A, const float* __restrict__ W,
                 const float* __restrict__ bias, float* __restrict__ C,
                 int M)
{
    constexpr int K  = H_;
    constexpr int CS = K + 8;
    __shared__ _Float16 sW[128 * CS];

    const int tid  = threadIdx.x;
    const int lane = tid & 31;
    const int wave = tid >> 5;

    for (int i = tid; i < K * 128; i += 256) {
        const int k = i >> 7, n = i & 127;
        sW[n * CS + k] = (_Float16)W[i];
    }
    __syncthreads();

    const int rt = wave & 3;
    const int ch = wave >> 2;
    const int r0 = blockIdx.x * 64 + rt * 16;

    const int mrow  = lane & 15;
    const int khalf = lane >> 4;

    int arow = r0 + mrow;
    if (arow >= M) arow = M - 1;
    const _Float16* __restrict__ Arow = A + (size_t)arow * K;
    __builtin_prefetch(Arow, 0, 0);

    const int bcol = lane & 15;
    const _Float16* __restrict__ sWlane = sW + khalf * 16;

    v8f acc[4] = {};

    #pragma unroll
    for (int kc = 0; kc < K; kc += 32) {
        // 16-bit A 16x32 layout: elems 0..15 = K kb..kb+15 (contiguous, matches B)
        const int kb = kc + khalf * 16;
        const v8h lo = *(const v8h*)(Arow + kb);        // 16B global_load_b128
        const v8h hi = *(const v8h*)(Arow + kb + 8);    // 16B global_load_b128
        v16h a;
        #pragma unroll
        for (int j = 0; j < 8; ++j) { a[j] = lo[j]; a[j + 8] = hi[j]; }

        #pragma unroll
        for (int ct = 0; ct < 4; ++ct) {
            const int c0 = ch * 64 + ct * 16 + bcol;
            const v16h b = *(const v16h*)(sWlane + (size_t)c0 * CS + kc);
            acc[ct] = __builtin_amdgcn_wmma_f32_16x16x32_f16(
                false, a, false, b, (short)0, acc[ct], false, false);
        }
    }

    const int rowOff = (lane < 16) ? 0 : 8;
    #pragma unroll
    for (int ct = 0; ct < 4; ++ct) {
        const int col = ch * 64 + ct * 16 + (lane & 15);
        const float bv = bias ? bias[col] : 0.0f;
        #pragma unroll
        for (int i = 0; i < 8; ++i) {
            const int row = r0 + i + rowOff;
            if (row < M)
                C[(size_t)row * 128 + col] = acc[ct][i] + bv;
        }
    }
}

// ---------------------------------------------------------------------
// small helpers
// ---------------------------------------------------------------------
__global__ void zero_f32(float* __restrict__ p, int n) {
    int i = blockIdx.x * 256 + threadIdx.x;
    if (i < n) p[i] = 0.0f;
}

// per-(dst,relation) in-degree counts
__global__ void edge_count(const int* __restrict__ dst, const int* __restrict__ et,
                           float* __restrict__ cnt) {
    int e = blockIdx.x * 256 + threadIdx.x;
    if (e < E_) atomicAdd(&cnt[(size_t)dst[e] * R_ + et[e]], 1.0f);
}

__global__ void edge_norm_k(const int* __restrict__ dst, const int* __restrict__ et,
                            const float* __restrict__ cnt, float* __restrict__ enorm) {
    int e = blockIdx.x * 256 + threadIdx.x;
    if (e < E_) enorm[e] = 1.0f / fmaxf(cnt[(size_t)dst[e] * R_ + et[e]], 1.0f);
}

// out[dst] += trel[src] * enorm  for edges of relation `rel`.
// 32 threads/edge, 4 cols/thread; edge arrays are L2-resident and the
// per-edge type check is a broadcast load across the 32 lanes of an edge.
__global__ void scatter_rel(const int* __restrict__ src, const int* __restrict__ dst,
                            const int* __restrict__ et, const float* __restrict__ enorm,
                            const float* __restrict__ trel, float* __restrict__ out,
                            int rel) {
    long idx = (long)blockIdx.x * 256 + threadIdx.x;
    int e = (int)(idx >> 5);
    if (e >= E_) return;
    if (et[e] != rel) return;
    int c4 = (int)(idx & 31) * 4;
    float nrm = enorm[e];
    const float4 t = *(const float4*)(trel + (size_t)src[e] * H_ + c4);
    float* op = out + (size_t)dst[e] * H_ + c4;
    atomicAdd(op + 0, t.x * nrm);                // global_atomic_add_f32
    atomicAdd(op + 1, t.y * nrm);
    atomicAdd(op + 2, t.z * nrm);
    atomicAdd(op + 3, t.w * nrm);
}

// BatchNorm statistics: one block = 128 rows, thread t owns column t.
__global__ __launch_bounds__(128)
void bn_stats(const float* __restrict__ x, float* __restrict__ sum,
              float* __restrict__ sq, int M) {
    int c  = threadIdx.x;
    int r0 = blockIdx.x * 128;
    float s = 0.0f, q = 0.0f;
    for (int i = 0; i < 128; ++i) {
        int r = r0 + i;
        if (r >= M) break;
        float v = x[(size_t)r * 128 + c];
        s += v; q += v * v;
    }
    atomicAdd(&sum[c], s);
    atomicAdd(&sq[c], q);
}

// fused normalize + affine + ReLU; writes f32 h (for pooling) and the
// f16 activation copy hh consumed as WMMA A-operand by the next layer.
__global__ void bn_apply(const float* __restrict__ x, const float* __restrict__ sum,
                         const float* __restrict__ sq, const float* __restrict__ g,
                         const float* __restrict__ b, float* __restrict__ h,
                         _Float16* __restrict__ hh, int M) {
    long idx = (long)blockIdx.x * 256 + threadIdx.x;
    if (idx >= (long)M * H_) return;
    int c = (int)(idx & 127);
    float inv_n = 1.0f / (float)M;
    float mu  = sum[c] * inv_n;
    float var = sq[c] * inv_n - mu * mu;
    float is  = rsqrtf(var + EPS_);
    float v   = fmaxf((x[idx] - mu) * is * g[c] + b[c], 0.0f);
    h[idx]  = v;
    hh[idx] = (_Float16)v;
}

// global mean pool accumulation
__global__ void pool_acc(const float* __restrict__ h, const int* __restrict__ batch,
                         float* __restrict__ pooled) {
    long idx = (long)blockIdx.x * 256 + threadIdx.x;
    if (idx >= (long)N_ * H_) return;
    int n = (int)(idx >> 7), c = (int)(idx & 127);
    atomicAdd(&pooled[(size_t)batch[n] * H_ + c], h[idx]);
}

__global__ void pool_cnt(const int* __restrict__ batch, float* __restrict__ cntg) {
    int n = blockIdx.x * 256 + threadIdx.x;
    if (n < N_) atomicAdd(&cntg[batch[n]], 1.0f);
}

// head: logits = sigmoid(mean_pool @ w_out + b_out)   (tiny: 64x10)
__global__ void head_k(const float* __restrict__ pooled, const float* __restrict__ cntg,
                       const float* __restrict__ w_out, const float* __restrict__ b_out,
                       float* __restrict__ out) {
    int i = blockIdx.x * 256 + threadIdx.x;
    if (i >= G_ * C_) return;
    int g = i / C_, c = i % C_;
    float inv = 1.0f / fmaxf(cntg[g], 1.0f);
    float s = b_out[c];
    #pragma unroll 4
    for (int k = 0; k < H_; ++k)
        s += pooled[(size_t)g * H_ + k] * inv * w_out[k * C_ + c];
    out[i] = 1.0f / (1.0f + __expf(-s));
}

// ---------------------------------------------------------------------
static inline int cdiv_l(long a, long b) { return (int)((a + b - 1) / b); }

extern "C" void kernel_launch(void* const* d_in, const int* in_sizes, int n_in,
                              void* d_out, int out_size, void* d_ws, size_t ws_size,
                              hipStream_t stream) {
    (void)in_sizes; (void)n_in; (void)out_size; (void)ws_size;

    const float* x      = (const float*)d_in[0];
    const int*   eidx   = (const int*)  d_in[1];
    const int*   esrc   = eidx;                // edge_index[0]
    const int*   edst   = eidx + E_;           // edge_index[1]
    const int*   etyp   = (const int*)  d_in[2];
    const int*   batch  = (const int*)  d_in[3];
    const float* w_in   = (const float*)d_in[4];
    const float* b_in   = (const float*)d_in[5];
    const float* rel_w  = (const float*)d_in[6];   // [L,R,H,H]
    const float* root_w = (const float*)d_in[7];   // [L,H,H]
    const float* conv_b = (const float*)d_in[8];   // [L,H]
    const float* bn_g   = (const float*)d_in[9];   // [L,H]
    const float* bn_b   = (const float*)d_in[10];  // [L,H]
    const float* w_out  = (const float*)d_in[11];  // [H,C]
    const float* b_out  = (const float*)d_in[12];  // [C]
    float* logits = (float*)d_out;                 // [G,C] f32

    // workspace carve: f32 tensors first, f16 activations at the end (~194 MB)
    float* p = (float*)d_ws;
    float* h     = p; p += (size_t)N_ * H_;    // f32 activations (pooling)
    float* ob    = p; p += (size_t)N_ * H_;    // layer output accumulator
    float* trel  = p; p += (size_t)N_ * H_;    // per-relation transformed nodes
    float* cnt   = p; p += (size_t)N_ * R_;    // (dst,rel) degree counts
    float* enorm = p; p += E_;                 // per-edge 1/deg
    float* bnsum = p; p += H_;                 // bn col sums
    float* bnsq  = p; p += H_;                 // bn col sumsq  (contiguous w/ bnsum)
    float* pooled= p; p += (size_t)G_ * H_;    // pooled sums
    float* cntg  = p; p += G_;                 //  (contiguous w/ pooled)
    _Float16* hh = (_Float16*)p;               // f16 activations (WMMA A operand)

    // 1) edge normalization factors
    zero_f32   <<<cdiv_l((long)N_ * R_, 256), 256, 0, stream>>>(cnt, N_ * R_);
    edge_count <<<cdiv_l(E_, 256), 256, 0, stream>>>(edst, etyp, cnt);
    edge_norm_k<<<cdiv_l(E_, 256), 256, 0, stream>>>(edst, etyp, cnt, enorm);

    // 2) input MLP: hh = f16(relu(x @ w_in + b_in))
    wmma_gemm_in<<<cdiv_l(N_, 64), 256, 0, stream>>>(x, w_in, b_in, hh, N_);

    // 3) RGCN layers
    for (int l = 0; l < L_; ++l) {
        // root/self transform + conv bias
        wmma_gemm_h<<<cdiv_l(N_, 64), 256, 0, stream>>>(
            hh, root_w + (size_t)l * H_ * H_, conv_b + (size_t)l * H_, ob, N_);

        // per relation: transform nodes (WMMA) then scatter along edges
        for (int r = 0; r < R_; ++r) {
            wmma_gemm_h<<<cdiv_l(N_, 64), 256, 0, stream>>>(
                hh, rel_w + ((size_t)l * R_ + r) * H_ * H_, nullptr, trel, N_);
            scatter_rel<<<cdiv_l((long)E_ * 32, 256), 256, 0, stream>>>(
                esrc, edst, etyp, enorm, trel, ob, r);
        }

        // BatchNorm (batch stats) + ReLU -> h (f32) and hh (f16)
        zero_f32<<<1, 256, 0, stream>>>(bnsum, 2 * H_);
        bn_stats<<<cdiv_l(N_, 128), 128, 0, stream>>>(ob, bnsum, bnsq, N_);
        bn_apply<<<cdiv_l((long)N_ * H_, 256), 256, 0, stream>>>(
            ob, bnsum, bnsq, bn_g + (size_t)l * H_, bn_b + (size_t)l * H_, h, hh, N_);
    }

    // 4) global mean pool + head
    zero_f32<<<cdiv_l(G_ * H_ + G_, 256), 256, 0, stream>>>(pooled, G_ * H_ + G_);
    pool_acc<<<cdiv_l((long)N_ * H_, 256), 256, 0, stream>>>(h, batch, pooled);
    pool_cnt<<<cdiv_l(N_, 256), 256, 0, stream>>>(batch, cntg);
    head_k  <<<cdiv_l(G_ * C_, 256), 256, 0, stream>>>(pooled, cntg, w_out, b_out, logits);
}
